// Attention_27719718928546
// MI455X (gfx1250) — matrix operand order repaired and checked by reading, not compiled, not run on hardware
//
#include <hip/hip_runtime.h>
#include <hip/hip_bf16.h>

#define SEQ 2048
#define HID 4096
#define NH  32
#define NKV 8
#define HD  128
#define QD  (NH*HD)    // 4096
#define KVD (NKV*HD)   // 1024

typedef __attribute__((ext_vector_type(16))) __bf16          v16bf;
typedef __attribute__((ext_vector_type(16))) unsigned short  v16us;
typedef __attribute__((ext_vector_type(8)))  float           v8f;
typedef __attribute__((ext_vector_type(4)))  int             v4i;

union V16 { v16us u; v16bf b; uint4 q[2]; };

// CDNA5 async global->LDS path (ASYNCcnt), guarded so either toolchain compiles.
#if defined(__HIP_DEVICE_COMPILE__) && __has_builtin(__builtin_amdgcn_global_load_async_to_lds_b128)
#define ASYNC_CP_DEV 1
#endif

#ifdef ASYNC_CP_DEV
typedef __attribute__((address_space(1))) v4i* gv4ip;   // global int4*
typedef __attribute__((address_space(3))) v4i* lv4ip;   // LDS int4*
__device__ inline void async_cp16(void* lds, const void* g) {
  __builtin_amdgcn_global_load_async_to_lds_b128((gv4ip)g, (lv4ip)lds, 0, 0);
}
__device__ inline void wait_async_le4() {
#if __has_builtin(__builtin_amdgcn_s_wait_asynccnt)
  __builtin_amdgcn_s_wait_asynccnt(4);
#else
  asm volatile("s_wait_asynccnt 0x4" ::: "memory");
#endif
}
__device__ inline void wait_async_0() {
#if __has_builtin(__builtin_amdgcn_s_wait_asynccnt)
  __builtin_amdgcn_s_wait_asynccnt(0);
#else
  asm volatile("s_wait_asynccnt 0x0" ::: "memory");
#endif
}
#endif

__device__ inline unsigned short f2bf(float f) {
  unsigned u = __float_as_uint(f);
  u += 0x7fffu + ((u >> 16) & 1u);           // round-to-nearest-even
  return (unsigned short)(u >> 16);
}

// Build a 16-element bf16 fragment from two contiguous 16-byte chunks.
__device__ inline v16bf frag_ab(const unsigned short* lo, const unsigned short* hi) {
  V16 t;
  t.q[0] = *(const uint4*)lo;
  t.q[1] = *(const uint4*)hi;
  return t.b;
}

__device__ inline v8f zero8() {
  v8f z;
#pragma unroll
  for (int i = 0; i < 8; i++) z[i] = 0.f;
  return z;
}

// ---------------- elementwise f32 -> bf16 ----------------
__global__ void k_f32_to_bf16(const float* __restrict__ in,
                              unsigned short* __restrict__ out, size_t n) {
  size_t i = (size_t)blockIdx.x * blockDim.x + threadIdx.x;
  size_t stride = (size_t)gridDim.x * blockDim.x;
  for (; i < n; i += stride) out[i] = f2bf(in[i]);
}

// ---------------- transpose-convert: in (K x N, f32) -> out (N x K, bf16) --
__global__ __launch_bounds__(256) void k_transpose_bf16(
    const float* __restrict__ in, unsigned short* __restrict__ out, int K, int N) {
  __shared__ float t[32][33];
  const int k0 = blockIdx.y * 32;
  const int n0 = blockIdx.x * 32;
  const int tx = threadIdx.x & 31, ty = threadIdx.x >> 5;   // 32 x 8
#pragma unroll
  for (int i = 0; i < 4; i++)
    t[ty + i*8][tx] = in[(size_t)(k0 + ty + i*8) * N + n0 + tx];
  __syncthreads();
#pragma unroll
  for (int i = 0; i < 4; i++)
    out[(size_t)(n0 + ty + i*8) * K + k0 + tx] = f2bf(t[tx][ty + i*8]);
}

// ---------------- bf16 GEMM: C(f32, MxN) = A(MxK) x Bt(NxK)^T -------------
// Block tile 128x128, K-step 32, 256 threads = 8 waves, wave tile 32x64.
// Double-buffered LDS; async global->LDS when available.
__global__ __launch_bounds__(256) void k_gemm_bf16(
    const unsigned short* __restrict__ A, const unsigned short* __restrict__ Bt,
    float* __restrict__ C, int M, int N, int K) {
  __shared__ unsigned short As[2][128][48];
  __shared__ unsigned short Bs[2][128][48];

  const int tid  = threadIdx.x;
  const int lane = tid & 31, wid = tid >> 5;
  const int half = lane >> 4, l16 = lane & 15;
  const int row0 = blockIdx.y * 128, col0 = blockIdx.x * 128;
  const int arow = (wid >> 1) * 32;
  const int bcol = (wid & 1) * 64;
  const int lr = tid >> 1;            // 0..127 (row within A/Bt tile)
  const int lc = (tid & 1) * 16;      // 0 or 16

  const unsigned short* ga = A  + (size_t)(row0 + lr) * K + lc;
  const unsigned short* gb = Bt + (size_t)(col0 + lr) * K + lc;

  v8f acc[2][4];
#pragma unroll
  for (int mi = 0; mi < 2; mi++)
#pragma unroll
    for (int ni = 0; ni < 4; ni++) acc[mi][ni] = zero8();

  const int nk = K >> 5;

#ifdef ASYNC_CP_DEV
  async_cp16(&As[0][lr][lc], ga);       async_cp16(&As[0][lr][lc + 8], ga + 8);
  async_cp16(&Bs[0][lr][lc], gb);       async_cp16(&Bs[0][lr][lc + 8], gb + 8);
#endif

  for (int t = 0; t < nk; ++t) {
    const int cur = t & 1;
#ifdef ASYNC_CP_DEV
    if (t + 1 < nk) {
      const int nxt = cur ^ 1;
      const unsigned short* pa = ga + (size_t)(t + 1) * 32;
      const unsigned short* pb = gb + (size_t)(t + 1) * 32;
      async_cp16(&As[nxt][lr][lc], pa); async_cp16(&As[nxt][lr][lc + 8], pa + 8);
      async_cp16(&Bs[nxt][lr][lc], pb); async_cp16(&Bs[nxt][lr][lc + 8], pb + 8);
      wait_async_le4();                 // current tile's 4 ops complete (in-order)
    } else {
      wait_async_0();
    }
#else
    {
      const unsigned short* pa = ga + (size_t)t * 32;
      const unsigned short* pb = gb + (size_t)t * 32;
      uint4 a0 = *(const uint4*)pa, a1 = *(const uint4*)(pa + 8);
      uint4 b0 = *(const uint4*)pb, b1 = *(const uint4*)(pb + 8);
      *(uint4*)&As[cur][lr][lc]     = a0;
      *(uint4*)&As[cur][lr][lc + 8] = a1;
      *(uint4*)&Bs[cur][lr][lc]     = b0;
      *(uint4*)&Bs[cur][lr][lc + 8] = b1;
    }
#endif
    __syncthreads();

    v16bf af[2], bfv[4];
#pragma unroll
    for (int mi = 0; mi < 2; mi++) {
      const unsigned short* p = &As[cur][arow + mi*16 + l16][0];
      af[mi] = frag_ab(p + 8*half, p + 16 + 8*half);          // A lane layout
    }
#pragma unroll
    for (int ni = 0; ni < 4; ni++) {
      const unsigned short* p = &Bs[cur][bcol + ni*16 + l16][0];
      bfv[ni] = frag_ab(p + 8*half, p + 16 + 8*half);        // B lane layout
    }
#pragma unroll
    for (int mi = 0; mi < 2; mi++)
#pragma unroll
      for (int ni = 0; ni < 4; ni++)
        acc[mi][ni] = __builtin_amdgcn_wmma_f32_16x16x32_bf16(
            false, af[mi], false, bfv[ni], (short)0, acc[mi][ni], false, false);
    __syncthreads();
  }

#pragma unroll
  for (int mi = 0; mi < 2; mi++)
#pragma unroll
    for (int ni = 0; ni < 4; ni++) {
      unsigned idx = (unsigned)((row0 + arow + mi*16 + half*8) * N +
                                 col0 + bcol + ni*16 + l16);
#pragma unroll
      for (int i = 0; i < 8; i++) C[idx + (unsigned)i * (unsigned)N] = acc[mi][ni][i];
    }
}

// ---------------- RMSNorm + RoPE, f32 -> bf16 repacked [head][seq][128] ----
__global__ __launch_bounds__(128) void k_rms_rope(
    const float* __restrict__ in, const float* __restrict__ w,
    const float* __restrict__ cosT, const float* __restrict__ sinT,
    const int* __restrict__ pos, unsigned short* __restrict__ out,
    int rowStride, float scale) {
  __shared__ float sh[128];
  const int s = blockIdx.x, h = blockIdx.y, d = threadIdx.x;
  const float* row = in + (size_t)s * rowStride + (size_t)h * HD;
  float x = row[d];
  sh[d] = x * x;
  __syncthreads();
#pragma unroll
  for (int off = 64; off > 0; off >>= 1) {
    if (d < off) sh[d] += sh[d + off];
    __syncthreads();
  }
  float ssq = sh[0];
  __syncthreads();
  float xn = x * rsqrtf(ssq * (1.f / HD) + 1e-6f) * w[d];
  sh[d] = xn;
  __syncthreads();
  float rot = (d < 64) ? -sh[d + 64] : sh[d - 64];
  int p = pos[s];
  float c  = cosT[(size_t)p * HD + d];
  float sn = sinT[(size_t)p * HD + d];
  out[((size_t)h * SEQ + s) * HD + d] = f2bf((xn * c + rot * sn) * scale);
}

// ---------------- V repack: f32 [s][kvh*128+d] -> bf16 [kvh][d][s] --------
__global__ __launch_bounds__(256) void k_repack_v(const float* __restrict__ vf,
                                                  unsigned short* __restrict__ vb) {
  __shared__ float t[32][33];
  const int kvh = blockIdx.z;
  const int s0 = blockIdx.x * 32;
  const int d0 = blockIdx.y * 32;
  const int tx = threadIdx.x & 31, ty = threadIdx.x >> 5;
#pragma unroll
  for (int i = 0; i < 4; i++)
    t[ty + i*8][tx] = vf[(size_t)(s0 + ty + i*8) * KVD + kvh*HD + d0 + tx];
  __syncthreads();
#pragma unroll
  for (int i = 0; i < 4; i++)
    vb[((size_t)kvh*HD + d0 + ty + i*8) * SEQ + s0 + tx] = f2bf(t[tx][ty + i*8]);
}

// ---------------- flash attention per head (scale pre-folded into Q) -----
// Q: [h][s][d] bf16, K: [kvh][s][d] bf16, V: [kvh][d][s] bf16 (pre-transposed)
__global__ __launch_bounds__(256) void k_attn(
    const unsigned short* __restrict__ qb, const unsigned short* __restrict__ kb,
    const unsigned short* __restrict__ vb, unsigned short* __restrict__ ob) {
  __shared__ unsigned short Ks[32][136];      // [key][d], padded
  __shared__ unsigned short Vt[128][40];      // [d][key], padded
  __shared__ unsigned short Ps[8][16][40];    // per-wave P staging, padded

  const int tid  = threadIdx.x;
  const int lane = tid & 31, wid = tid >> 5;
  const int half = lane >> 4, l16 = lane & 15;
  const int h    = blockIdx.y;
  const int kvh  = h >> 2;                    // GQA: 4 Q heads per KV head
  const int q0   = blockIdx.x * 128;
  const int qrow = q0 + wid * 16;

  const unsigned short* qh = qb + (size_t)h   * SEQ * HD;
  const unsigned short* kh = kb + (size_t)kvh * SEQ * HD;
  const unsigned short* vh = vb + (size_t)kvh * SEQ * HD;  // [d][s]

  // Q strip (16 rows x 128) as 4 A-fragments
  v16bf qfr[4];
#pragma unroll
  for (int kc = 0; kc < 4; kc++) {
    const unsigned short* p = qh + (size_t)(qrow + l16) * HD + kc * 32;
    qfr[kc] = frag_ab(p + 8*half, p + 16 + 8*half);
  }

  float m[8], l[8];
  v8f o[8];
#pragma unroll
  for (int i = 0; i < 8; i++) { m[i] = -3.0e38f; l[i] = 0.f; o[i] = zero8(); }

  const int kr = tid >> 3;            // 0..31  K tile row (key)
  const int kc2 = (tid & 7) * 16;     // 0..112 K tile col (d)
  const int vr = tid >> 1;            // 0..127 V tile row (d)
  const int vc = (tid & 1) * 16;      // 0/16   V tile col (key)
  const int nj = q0 / 32 + 4;         // causal: keys <= q0+127

  for (int jt = 0; jt < nj; jt++) {
    const int jb = jt * 32;
    const unsigned short* gk = kh + (size_t)(jb + kr) * HD + kc2;
    const unsigned short* gv = vh + (size_t)vr * SEQ + jb + vc;
#ifdef ASYNC_CP_DEV
    async_cp16(&Ks[kr][kc2], gk);     async_cp16(&Ks[kr][kc2 + 8], gk + 8);
    async_cp16(&Vt[vr][vc], gv);      async_cp16(&Vt[vr][vc + 8], gv + 8);
    wait_async_0();
#else
    {
      uint4 x0 = *(const uint4*)gk, x1 = *(const uint4*)(gk + 8);
      *(uint4*)&Ks[kr][kc2]     = x0;
      *(uint4*)&Ks[kr][kc2 + 8] = x1;
      uint4 y0 = *(const uint4*)gv, y1 = *(const uint4*)(gv + 8);
      *(uint4*)&Vt[vr][vc]     = y0;
      *(uint4*)&Vt[vr][vc + 8] = y1;
    }
#endif
    __syncthreads();

    // scores: two 16x16 tiles over key sub-tiles
    v8f c1 = zero8(), c2 = zero8();
#pragma unroll
    for (int kc = 0; kc < 4; kc++) {
      const unsigned short* p1 = &Ks[l16][kc * 32];
      const unsigned short* p2 = &Ks[16 + l16][kc * 32];
      v16bf b1 = frag_ab(p1 + 8*half, p1 + 16 + 8*half);
      v16bf b2 = frag_ab(p2 + 8*half, p2 + 16 + 8*half);
      c1 = __builtin_amdgcn_wmma_f32_16x16x32_bf16(false, qfr[kc], false, b1, (short)0, c1, false, false);
      c2 = __builtin_amdgcn_wmma_f32_16x16x32_bf16(false, qfr[kc], false, b2, (short)0, c2, false, false);
    }

    float fs[8];
#pragma unroll
    for (int i = 0; i < 8; i++) {
      int qpos = qrow + i + half * 8;
      float s1 = c1[i] + (((jb + l16)      <= qpos) ? 0.f : -1.0e9f);
      float s2 = c2[i] + (((jb + 16 + l16) <= qpos) ? 0.f : -1.0e9f);
      float mx = fmaxf(s1, s2);
      mx = fmaxf(mx, __shfl_xor(mx, 1, 16));
      mx = fmaxf(mx, __shfl_xor(mx, 2, 16));
      mx = fmaxf(mx, __shfl_xor(mx, 4, 16));
      mx = fmaxf(mx, __shfl_xor(mx, 8, 16));
      float mn  = fmaxf(m[i], mx);
      float p1v = __expf(s1 - mn);
      float p2v = __expf(s2 - mn);
      float rs  = p1v + p2v;
      rs += __shfl_xor(rs, 1, 16);
      rs += __shfl_xor(rs, 2, 16);
      rs += __shfl_xor(rs, 4, 16);
      rs += __shfl_xor(rs, 8, 16);
      float f = __expf(m[i] - mn);
      l[i] = l[i] * f + rs;
      m[i] = mn;
      fs[i] = f;
      int row = i + half * 8;
      Ps[wid][row][l16]      = f2bf(p1v);
      Ps[wid][row][16 + l16] = f2bf(p2v);
    }

#pragma unroll
    for (int t = 0; t < 8; t++)
#pragma unroll
      for (int i = 0; i < 8; i++) o[t][i] *= fs[i];

    const unsigned short* pp = &Ps[wid][l16][0];
    v16bf pfr = frag_ab(pp + 8*half, pp + 16 + 8*half);
#pragma unroll
    for (int t = 0; t < 8; t++) {
      const unsigned short* vp = &Vt[t * 16 + l16][0];
      v16bf vfr = frag_ab(vp + 8*half, vp + 16 + 8*half);
      o[t] = __builtin_amdgcn_wmma_f32_16x16x32_bf16(false, pfr, false, vfr, (short)0, o[t], false, false);
    }
    __syncthreads();
  }

#pragma unroll
  for (int i = 0; i < 8; i++) l[i] = 1.f / l[i];
#pragma unroll
  for (int t = 0; t < 8; t++)
#pragma unroll
    for (int i = 0; i < 8; i++) {
      int r = qrow + i + half * 8;
      int c = h * HD + t * 16 + l16;
      ob[(size_t)r * QD + c] = f2bf(o[t][i] * l[i]);
    }
}

// ---------------------------------------------------------------------------
extern "C" void kernel_launch(void* const* d_in, const int* in_sizes, int n_in,
                              void* d_out, int out_size, void* d_ws, size_t ws_size,
                              hipStream_t stream) {
  const float* x    = (const float*)d_in[0];
  const int*   pos  = (const int*)d_in[1];
  const float* cosT = (const float*)d_in[2];
  const float* sinT = (const float*)d_in[3];
  // d_in[4] = attn_mask (exact causal 0/-1e9) — applied analytically in k_attn
  const float* Wq = (const float*)d_in[5];
  const float* Wk = (const float*)d_in[6];
  const float* Wv = (const float*)d_in[7];
  const float* Wo = (const float*)d_in[8];
  const float* qw = (const float*)d_in[9];
  const float* kw = (const float*)d_in[10];
  float* out = (float*)d_out;

  char* base = (char*)d_ws;
  size_t off = 0;
  auto alloc = [&](size_t elems, size_t esz) {
    char* p = base + off;
    off += (elems * esz + 255) & ~(size_t)255;
    return p;
  };
  unsigned short* xb  = (unsigned short*)alloc((size_t)SEQ * HID, 2);
  unsigned short* wqt = (unsigned short*)alloc((size_t)QD  * HID, 2);  // Wq^T
  unsigned short* wkt = (unsigned short*)alloc((size_t)KVD * HID, 2);  // Wk^T
  unsigned short* wvt = (unsigned short*)alloc((size_t)KVD * HID, 2);  // Wv^T
  unsigned short* wot = (unsigned short*)alloc((size_t)HID * QD, 2);   // Wo^T
  float* qf = (float*)alloc((size_t)SEQ * QD, 4);
  float* kf = (float*)alloc((size_t)SEQ * KVD, 4);
  float* vf = (float*)alloc((size_t)SEQ * KVD, 4);
  unsigned short* qbh = (unsigned short*)alloc((size_t)NH  * SEQ * HD, 2);
  unsigned short* kbh = (unsigned short*)alloc((size_t)NKV * SEQ * HD, 2);
  unsigned short* vbh = (unsigned short*)alloc((size_t)NKV * SEQ * HD, 2);
  unsigned short* obh = (unsigned short*)alloc((size_t)SEQ * QD, 2);

  k_f32_to_bf16<<<2048, 256, 0, stream>>>(x, xb, (size_t)SEQ * HID);
  k_transpose_bf16<<<dim3(QD  / 32, HID / 32), 256, 0, stream>>>(Wq, wqt, HID, QD);
  k_transpose_bf16<<<dim3(KVD / 32, HID / 32), 256, 0, stream>>>(Wk, wkt, HID, KVD);
  k_transpose_bf16<<<dim3(KVD / 32, HID / 32), 256, 0, stream>>>(Wv, wvt, HID, KVD);
  k_transpose_bf16<<<dim3(HID / 32, QD  / 32), 256, 0, stream>>>(Wo, wot, QD, HID);

  k_gemm_bf16<<<dim3(QD  / 128, SEQ / 128), 256, 0, stream>>>(xb, wqt, qf, SEQ, QD,  HID);
  k_gemm_bf16<<<dim3(KVD / 128, SEQ / 128), 256, 0, stream>>>(xb, wkt, kf, SEQ, KVD, HID);
  k_gemm_bf16<<<dim3(KVD / 128, SEQ / 128), 256, 0, stream>>>(xb, wvt, vf, SEQ, KVD, HID);

  const float scale = 0.0883883476483184f;  // 1/sqrt(128), folded into Q
  k_rms_rope<<<dim3(SEQ, NH),  128, 0, stream>>>(qf, qw, cosT, sinT, pos, qbh, QD,  scale);
  k_rms_rope<<<dim3(SEQ, NKV), 128, 0, stream>>>(kf, kw, cosT, sinT, pos, kbh, KVD, 1.0f);
  k_repack_v<<<dim3(SEQ / 32, HD / 32, NKV), 256, 0, stream>>>(vf, vbh);

  k_attn<<<dim3(SEQ / 128, NH), 256, 0, stream>>>(qbh, kbh, vbh, obh);

  k_gemm_bf16<<<dim3(HID / 128, SEQ / 128), 256, 0, stream>>>(obh, wot, out, SEQ, HID, QD);

  (void)in_sizes; (void)n_in; (void)out_size; (void)ws_size;
}
